// ExpertMoE_40510131536110
// MI455X (gfx1250) — compile-verified, matrix operand order checked by reading, not run on hardware
//
#include <hip/hip_runtime.h>

// ---------------- problem constants ----------------
#define BS   16384      // B*S tokens
#define DD   1024       // model dim
#define EE   8          // experts
#define HH   768        // expert hidden
#define OO   1024       // output dim

typedef __attribute__((ext_vector_type(16))) __bf16       v16bf;
typedef __attribute__((ext_vector_type(8)))  __bf16       v8bf;
typedef __attribute__((ext_vector_type(8)))  float        v8f;
typedef __attribute__((ext_vector_type(8)))  unsigned int v8u;

// load a 16-element bf16 operand chunk pair (two 16B loads, no lane shuffles)
__device__ __forceinline__ v16bf ld16(const __bf16* p) {
    v8bf lo = *(const v8bf*)(p);
    v8bf hi = *(const v8bf*)(p + 16);
    return __builtin_shufflevector(lo, hi,
        0, 1, 2, 3, 4, 5, 6, 7, 8, 9, 10, 11, 12, 13, 14, 15);
}

// pack a float into a dword holding its bf16 value in both halves
__device__ __forceinline__ unsigned packbf(float f) {
    __bf16 b = (__bf16)f;
    unsigned short us = __builtin_bit_cast(unsigned short, b);
    return (unsigned)us * 0x10001u;
}

// scale a bf16 operand vector by a packed bf16 splat: 8x v_pk_mul_bf16,
// second source is ONE shared VGPR (avoids the compiler materializing
// 8 redundant splat registers per expert).
__device__ __forceinline__ v16bf scale16(v16bf a, unsigned gpack) {
    v8u au = __builtin_bit_cast(v8u, a);
    v8u r;
#pragma unroll
    for (int i = 0; i < 8; ++i) {
        unsigned t;
        asm("v_pk_mul_bf16 %0, %1, %2" : "=v"(t) : "v"(au[i]), "v"(gpack));
        r[i] = t;
    }
    return __builtin_bit_cast(v16bf, r);
}

// ---------------- converters ----------------
__global__ void k_cvt_x(const float* __restrict__ x, __bf16* __restrict__ xh) {
    long i = (long)blockIdx.x * blockDim.x + threadIdx.x;   // BS*DD threads
    xh[i] = (__bf16)x[i];
}

// We [E][D][H] f32  ->  WehT [E][H][D] bf16 (transposed: B-operand cols contiguous)
__global__ void k_cvt_We(const float* __restrict__ We, __bf16* __restrict__ WehT) {
    long i = (long)blockIdx.x * blockDim.x + threadIdx.x;   // E*D*H threads
    int h = (int)(i % HH);
    long t = i / HH;
    int d = (int)(t % DD);
    int e = (int)(t / DD);
    WehT[((long)e * HH + h) * DD + d] = (__bf16)We[i];
}

// Wo [H][O] f32 -> WoT [O][H] bf16
__global__ void k_cvt_Wo(const float* __restrict__ Wo, __bf16* __restrict__ WoT) {
    long i = (long)blockIdx.x * blockDim.x + threadIdx.x;   // H*O threads
    int o = (int)(i % OO);
    int h = (int)(i / OO);
    WoT[(long)o * HH + h] = (__bf16)Wo[i];
}

// ---------------- gate: softmax(x @ Wg + bg) ----------------
// one wave32 per token; 8 waves per 256-thread block
__global__ void k_gate(const float* __restrict__ x, const float* __restrict__ Wg,
                       const float* __restrict__ bg, float* __restrict__ gate) {
    const int lane  = threadIdx.x & 31;
    const int wave  = threadIdx.x >> 5;
    const int token = blockIdx.x * 8 + wave;

    float acc[EE];
#pragma unroll
    for (int e = 0; e < EE; ++e) acc[e] = 0.0f;

    const float* xr = x + (long)token * DD;
    for (int d = lane; d < DD; d += 32) {
        float xv = xr[d];
        const float* wr = Wg + (long)d * EE;
#pragma unroll
        for (int e = 0; e < EE; ++e) acc[e] = fmaf(xv, wr[e], acc[e]);
    }
#pragma unroll
    for (int e = 0; e < EE; ++e) {
#pragma unroll
        for (int off = 16; off > 0; off >>= 1)
            acc[e] += __shfl_xor(acc[e], off, 32);
    }
    if (lane == 0) {
        float m = -1e30f;
#pragma unroll
        for (int e = 0; e < EE; ++e) { acc[e] += bg[e]; m = fmaxf(m, acc[e]); }
        float s = 0.0f;
#pragma unroll
        for (int e = 0; e < EE; ++e) { acc[e] = __expf(acc[e] - m); s += acc[e]; }
        float inv = 1.0f / s;
#pragma unroll
        for (int e = 0; e < EE; ++e) gate[(long)token * EE + e] = acc[e] * inv;
    }
}

// ---------------- fused experts + gated aggregation ----------------
// Wave tile: 32 tokens x 64 H. Gate pre-scales A rows (bf16), so all experts
// accumulate into ONE accumulator set; each B tile feeds two WMMAs.
// All 8 waves in a block share the same N-tile -> B reused through WGP$.
__global__ void __launch_bounds__(256, 1)
k_moe(const __bf16* __restrict__ xh, const __bf16* __restrict__ WehT,
      const float* __restrict__ be, const float* __restrict__ gate,
      __bf16* __restrict__ aggH) {
    const int lane  = threadIdx.x & 31;
    const int wave  = threadIdx.x >> 5;
    const int NT    = HH / 64;                          // 12
    const int ntile = blockIdx.x % NT;
    const int mtile = (blockIdx.x / NT) * 8 + wave;     // 512 tiles of 32 tokens
    const int m0  = mtile * 32;
    const int h0  = ntile * 64;
    const int row = lane & 15;              // A row (per half) / B,C,D column
    const int hiK = (lane >> 4) * 8;        // K sub-offset for A/B packing
    const int hi8 = hiK;                    // +8 row offset for C/D upper half

    v8f agg[8];                             // [n][mhalf] -> agg[2n+mh]
#pragma unroll
    for (int i = 0; i < 8; ++i) agg[i] = v8f{};

    // gate scale factors: one packed dword per (expert, row-half) -> 16 VGPRs
    unsigned gp0[EE], gp1[EE];
#pragma unroll
    for (int e = 0; e < EE; ++e) {
        gp0[e] = packbf(gate[(long)(m0 + row) * EE + e]);
        gp1[e] = packbf(gate[(long)(m0 + 16 + row) * EE + e]);
    }

    const __bf16* A0 = xh + (long)(m0 + row) * DD;
    const __bf16* A1 = A0 + (long)16 * DD;

#pragma unroll 1
    for (int k0 = 0; k0 < DD; k0 += 32) {
        v16bf a0 = ld16(A0 + k0 + hiK);     // shared by all 8 experts
        v16bf a1 = ld16(A1 + k0 + hiK);
        __builtin_prefetch(A0 + k0 + 32, 0, 1);
        __builtin_prefetch(A1 + k0 + 32, 0, 1);
#pragma unroll
        for (int e = 0; e < EE; ++e) {
            v16bf a0e = scale16(a0, gp0[e]);
            v16bf a1e = scale16(a1, gp1[e]);
            const __bf16* Bb = WehT + ((long)e * HH + h0 + row) * DD + k0 + hiK;
#pragma unroll
            for (int n = 0; n < 4; ++n) {
                v16bf b = ld16(Bb + (long)n * 16 * DD);
                agg[2 * n + 0] = __builtin_amdgcn_wmma_f32_16x16x32_bf16(
                    false, a0e, false, b, (short)0, agg[2 * n + 0], false, false);
                agg[2 * n + 1] = __builtin_amdgcn_wmma_f32_16x16x32_bf16(
                    false, a1e, false, b, (short)0, agg[2 * n + 1], false, false);
            }
        }
    }

    // bias epilogue: agg += sum_e gate[row,e] * be[e,h]
#pragma unroll
    for (int e = 0; e < EE; ++e) {
        float gv0[8], gv1[8];
#pragma unroll
        for (int v = 0; v < 8; ++v) {
            gv0[v] = gate[(long)(m0 + v + hi8) * EE + e];
            gv1[v] = gate[(long)(m0 + 16 + v + hi8) * EE + e];
        }
#pragma unroll
        for (int n = 0; n < 4; ++n) {
            float bev = be[(long)e * HH + h0 + n * 16 + row];
#pragma unroll
            for (int v = 0; v < 8; ++v) {
                agg[2 * n + 0][v] = fmaf(gv0[v], bev, agg[2 * n + 0][v]);
                agg[2 * n + 1][v] = fmaf(gv1[v], bev, agg[2 * n + 1][v]);
            }
        }
    }

    // store aggregate tile as bf16 [token][H]
#pragma unroll
    for (int n = 0; n < 4; ++n) {
        int h = h0 + n * 16 + row;
#pragma unroll
        for (int v = 0; v < 8; ++v) {
            aggH[(long)(m0 + v + hi8) * HH + h]      = (__bf16)agg[2 * n + 0][v];
            aggH[(long)(m0 + 16 + v + hi8) * HH + h] = (__bf16)agg[2 * n + 1][v];
        }
    }
}

// ---------------- output GEMM: out = aggH @ Wo + bo ----------------
// Same 32x64 wave tile; waves in a block share the N-tile of WoT.
__global__ void __launch_bounds__(256, 1)
k_out(const __bf16* __restrict__ aggH, const __bf16* __restrict__ WoT,
      const float* __restrict__ bo, float* __restrict__ out) {
    const int lane  = threadIdx.x & 31;
    const int wave  = threadIdx.x >> 5;
    const int NT    = OO / 64;                          // 16
    const int ntile = blockIdx.x % NT;
    const int mtile = (blockIdx.x / NT) * 8 + wave;     // 512 tiles of 32 tokens
    const int m0  = mtile * 32;
    const int o0  = ntile * 64;
    const int row = lane & 15;
    const int hiK = (lane >> 4) * 8;
    const int hi8 = hiK;

    v8f acc[8];
#pragma unroll
    for (int i = 0; i < 8; ++i) acc[i] = v8f{};

    const __bf16* A0 = aggH + (long)(m0 + row) * HH;
    const __bf16* A1 = A0 + (long)16 * HH;

#pragma unroll 1
    for (int k0 = 0; k0 < HH; k0 += 32) {
        v16bf a0 = ld16(A0 + k0 + hiK);
        v16bf a1 = ld16(A1 + k0 + hiK);
        __builtin_prefetch(A0 + k0 + 32, 0, 1);
#pragma unroll
        for (int n = 0; n < 4; ++n) {
            const __bf16* Bcol = WoT + (long)(o0 + n * 16 + row) * HH + k0 + hiK;
            v16bf b = ld16(Bcol);
            acc[2 * n + 0] = __builtin_amdgcn_wmma_f32_16x16x32_bf16(
                false, a0, false, b, (short)0, acc[2 * n + 0], false, false);
            acc[2 * n + 1] = __builtin_amdgcn_wmma_f32_16x16x32_bf16(
                false, a1, false, b, (short)0, acc[2 * n + 1], false, false);
        }
    }

#pragma unroll
    for (int n = 0; n < 4; ++n) {
        int o = o0 + n * 16 + row;
        float bov = bo[o];
#pragma unroll
        for (int v = 0; v < 8; ++v) {
            out[(long)(m0 + v + hi8) * OO + o]      = acc[2 * n + 0][v] + bov;
            out[(long)(m0 + 16 + v + hi8) * OO + o] = acc[2 * n + 1][v] + bov;
        }
    }
}

// ---------------- launcher ----------------
extern "C" void kernel_launch(void* const* d_in, const int* in_sizes, int n_in,
                              void* d_out, int out_size, void* d_ws, size_t ws_size,
                              hipStream_t stream) {
    const float* x  = (const float*)d_in[0];
    const float* We = (const float*)d_in[1];
    const float* be = (const float*)d_in[2];
    const float* Wg = (const float*)d_in[3];
    const float* bg = (const float*)d_in[4];
    const float* Wo = (const float*)d_in[5];
    const float* bo = (const float*)d_in[6];
    float* out = (float*)d_out;

    char* ws = (char*)d_ws;
    size_t off = 0;
    float*  gate = (float*)(ws + off);  off += (size_t)BS * EE * sizeof(float);   // 0.5 MB
    __bf16* xh   = (__bf16*)(ws + off); off += (size_t)BS * DD * 2;               // 32 MB
    __bf16* WehT = (__bf16*)(ws + off); off += (size_t)EE * HH * DD * 2;          // 12 MB
    __bf16* WoT  = (__bf16*)(ws + off); off += (size_t)OO * HH * 2;               // 1.5 MB
    __bf16* aggH = (__bf16*)(ws + off); off += (size_t)BS * HH * 2;               // 24 MB

    // converters
    k_cvt_x <<<(BS * DD) / 256,      256, 0, stream>>>(x, xh);
    k_cvt_We<<<(EE * DD * HH) / 256, 256, 0, stream>>>(We, WehT);
    k_cvt_Wo<<<(HH * OO) / 256,      256, 0, stream>>>(Wo, WoT);

    // gates: 8 tokens per 256-thread block
    k_gate<<<BS / 8, 256, 0, stream>>>(x, Wg, bg, gate);

    // fused experts + gated aggregation:
    // (BS/32)/8 m-groups x (HH/64) n-tiles = 64*12 = 768 blocks
    k_moe<<<((BS / 32) / 8) * (HH / 64), 256, 0, stream>>>(xh, WehT, be, gate, aggH);

    // output GEMM: 64*16 = 1024 blocks
    k_out<<<((BS / 32) / 8) * (OO / 64), 256, 0, stream>>>(aggH, WoT, bo, out);
}
